// PointerDecoderSort_57578331570395
// MI455X (gfx1250) — matrix operand order repaired and checked by reading, not compiled, not run on hardware
//
#include <hip/hip_runtime.h>
#include <math.h>

typedef __attribute__((ext_vector_type(16))) _Float16 v16h;
typedef __attribute__((ext_vector_type(8)))  float    v8f;
using half_t = _Float16;

// ---------------- problem constants ----------------
#define B_SZ     4096
#define NTOK     100
#define DMODEL   128
#define DFF      256
#define MPAD     112          // 7 * 16 (M padding for 16x16 tiles)
#define NTHREADS 256
#define NWAVES   8

// ---------------- workspace layout (offsets in halves) ----------------
// Weights stored f16, pre-swizzled into WMMA B-fragment order:
//   half index o: i=o&15 (half within lane), lane=(o>>4)&31, frag=o>>9,
//   nt=frag%NT, kt=frag/NT, k=kt*32+((lane>>4)<<4)+i, n=nt*16+(lane&15)
#define OFF_VE   0            // ve_W2  64x128
#define OFF_WQ   8192         // Wq    128x128
#define OFF_WK   24576        // Wk
#define OFF_WV   40960        // Wv
#define OFF_WO   57344        // Wo
#define OFF_W1   73728        // ffn_W1 128x256
#define OFF_W2   106496       // ffn_W2 256x128
#define OFF_WKP  139264       // Wkp   128x128
#define OFF_QRY  155648       // queries (rank_emb@Wqp+bqp) f16 row-major [112][128]
#define OFF_POSQ 169984       // f32! zero-padded (ve_b2 + pos_emb) [112][128]
#define WS_HALVES (OFF_POSQ + MPAD * DMODEL * 2)   // ~400 KB

// ---------------- LDS layout (byte offsets) ----------------
#define L_H32   0             // h residual f32 [112][128]  57344
#define L_H16   57344         // h f16        [112][128]    28672
#define L_Q16   86016         // q f16 [112][128] (aliased by FFN f1 [112][256])
#define L_K16   114688        // k f16 [112][128]
#define L_VT16  143360        // v^T f16 [128][128] (aliased early by e16 [112][64])
#define L_SC32  176128        // f32 scratch [112][128]     57344
#define L_AT16  233472        // attn f16 [112][128]
#define L_AO16  262144        // attn-out f16 [112][128] (aliased later by keys)
#define L_PICK  290816        // 128 ints
#define L_BIAS  291328        // 1536 f32 staged biases / LN params
#define LDS_BYTES 297472

// bias block offsets (floats within L_BIAS)
#define BI_BQ   0
#define BI_BK   128
#define BI_BV   256
#define BI_BO   384
#define BI_FB1  512          // 256 entries
#define BI_FB2  768
#define BI_BKP  896
#define BI_L1G  1024
#define BI_L1B  1152
#define BI_L2G  1280
#define BI_L2B  1408

// B-fragment loader from pre-swizzled f16 weights in workspace
#define BWS(OFFv, NTv)                                                         \
  [&](int kt, int nt, int lane, float4* qq) {                                  \
    const float4* p =                                                          \
        (const float4*)(ws + (OFFv) + (((kt * (NTv) + nt) * 32 + lane) << 4)); \
    qq[0] = p[0]; qq[1] = p[1];                                                \
  }

// B-fragment loader from a row-major "B-transposed" source Bt[n][k]
// (covers K^T, V^T, keys^T: lane -> column n fixed, K contiguous)
#define BTR(BTp, LDBv)                                                         \
  [&](int kt, int nt, int lane, float4* qq) {                                  \
    int l = lane & 15, hi = lane >> 4;                                         \
    const float4* p =                                                          \
        (const float4*)((BTp) + (nt * 16 + l) * (LDBv) + kt * 32 + hi * 16);   \
    qq[0] = p[0]; qq[1] = p[1];                                                \
  }

// Block-cooperative GEMM over 16x16 output tiles using v_wmma_f32_16x16x32_f16.
// A: row-major f16 (LDS or global), contiguous-K A-frag layout (k = hi*16 + i) -> 2x b128 per lane.
// MT/NT/KT compile-time so tile decomposition and offsets constant-fold.
template <int MT, int NT, int KT, class BLoad, class Store>
__device__ __forceinline__ void gemm16(const half_t* A, int ldA, BLoad bload,
                                       Store store) {
  const int tid = threadIdx.x;
  const int wave = tid >> 5;
  const int lane = tid & 31;
  const int l = lane & 15;
  const int hi = lane >> 4;
  for (int t = wave; t < MT * NT; t += NWAVES) {
    const int mt = t / NT, nt = t % NT;
    v8f c = {};
#pragma unroll
    for (int kt = 0; kt < KT; ++kt) {
      union { v16h v; float4 q[2]; } a, b;
      const half_t* ap = A + (mt * 16 + l) * ldA + kt * 32 + hi * 16;
      a.q[0] = *(const float4*)(ap);        // halves 0..7  : K = kBase+hi*16+0..7
      a.q[1] = *(const float4*)(ap + 8);    // halves 8..15 : K = kBase+hi*16+8..15
      bload(kt, nt, lane, b.q);
      c = __builtin_amdgcn_wmma_f32_16x16x32_f16(false, a.v, false, b.v,
                                                 (short)0, c, false, false);
    }
#pragma unroll
    for (int j = 0; j < 8; ++j)
      store(mt * 16 + hi * 8 + j, nt * 16 + l, c[j]);  // row M, col N
  }
}

// ---------------- prep: f32 -> f16 swizzled weights + queries + posq --------
__global__ void __launch_bounds__(256) pds_prep(
    const float* __restrict__ rankE, const float* __restrict__ Wqp,
    const float* __restrict__ bqp, const float* __restrict__ veW2,
    const float* __restrict__ Wq, const float* __restrict__ Wk,
    const float* __restrict__ Wv, const float* __restrict__ Wo,
    const float* __restrict__ W1, const float* __restrict__ W2,
    const float* __restrict__ Wkp, const float* __restrict__ veB2,
    const float* __restrict__ posE, half_t* __restrict__ ws) {
  const float* srcs[8] = {veW2, Wq, Wk, Wv, Wo, W1, W2, Wkp};
  const int Ks[8] = {64, 128, 128, 128, 128, 128, 256, 128};
  const int Ns[8] = {128, 128, 128, 128, 128, 256, 128, 128};
  const int offs[8] = {OFF_VE, OFF_WQ, OFF_WK, OFF_WV,
                       OFF_WO, OFF_W1, OFF_W2, OFF_WKP};
  const int gstride = gridDim.x * blockDim.x;
  const int gtid = blockIdx.x * blockDim.x + threadIdx.x;
  for (int w = 0; w < 8; ++w) {
    const int total = Ks[w] * Ns[w];
    const int NT = Ns[w] >> 4;
    const float* src = srcs[w];
    for (int o = gtid; o < total; o += gstride) {
      int i = o & 15, lane = (o >> 4) & 31, f = o >> 9;
      int nt = f % NT, kt = f / NT;
      int k = kt * 32 + ((lane >> 4) << 4) + i;
      int n = nt * 16 + (lane & 15);
      ws[offs[w] + o] = (half_t)src[k * Ns[w] + n];
    }
  }
  // queries[r][d] = bqp[d] + sum_k rank_emb[r][k]*Wqp[k][d], zero-padded rows
  for (int o = gtid; o < MPAD * DMODEL; o += gstride) {
    int r = o >> 7, d = o & 127;
    float acc = 0.f;
    if (r < NTOK) {
      acc = bqp[d];
      for (int k = 0; k < DMODEL; ++k)
        acc += rankE[r * DMODEL + k] * Wqp[k * DMODEL + d];
    }
    ws[OFF_QRY + o] = (half_t)acc;
  }
  // posq[r][d] = ve_b2[d] + pos_emb[r][d], zero rows r>=100 (keeps pads exact 0)
  float* posq = (float*)(ws + OFF_POSQ);
  for (int o = gtid; o < MPAD * DMODEL; o += gstride) {
    int r = o >> 7, d = o & 127;
    posq[o] = (r < NTOK) ? (veB2[d] + posE[r * DMODEL + d]) : 0.f;
  }
}

// ---------------- main: one workgroup per batch element ----------------
__global__ void __launch_bounds__(NTHREADS, 1) pds_main(
    const float* __restrict__ x, const float* __restrict__ veW1,
    const float* __restrict__ veB1, const float* __restrict__ bq,
    const float* __restrict__ bk, const float* __restrict__ bv,
    const float* __restrict__ bo, const float* __restrict__ fb1,
    const float* __restrict__ fb2, const float* __restrict__ ln1g,
    const float* __restrict__ ln1b, const float* __restrict__ ln2g,
    const float* __restrict__ ln2b, const float* __restrict__ bkp,
    const half_t* __restrict__ ws, float* __restrict__ out) {
  extern __shared__ char smem[];
  float*  h32  = (float*)(smem + L_H32);
  half_t* h16  = (half_t*)(smem + L_H16);
  half_t* q16  = (half_t*)(smem + L_Q16);
  half_t* k16  = (half_t*)(smem + L_K16);
  half_t* vt16 = (half_t*)(smem + L_VT16);
  half_t* e16  = (half_t*)(smem + L_VT16);  // alias (dead before vt16 lives)
  float*  sc32 = (float*)(smem + L_SC32);
  half_t* at16 = (half_t*)(smem + L_AT16);
  half_t* ao16 = (half_t*)(smem + L_AO16);
  half_t* key16 = (half_t*)(smem + L_AO16); // alias (dead before keys live)
  half_t* f116 = (half_t*)(smem + L_Q16);   // alias, spans q16+k16 (112x256)
  int*    pick = (int*)(smem + L_PICK);
  float*  lb   = (float*)(smem + L_BIAS);

  const int tid = threadIdx.x;
  const int b = blockIdx.x;
  const float* posq = (const float*)(ws + OFF_POSQ);

  // L2-resident weights: warm the fragment stream early
  if (tid < 64) __builtin_prefetch(ws + ((size_t)tid * (WS_HALVES / 64)), 0, 3);

  // ---- stage 0: stage biases / LN params into LDS (used from stage 3 on)
  for (int i = tid; i < DMODEL; i += NTHREADS) {
    lb[BI_BQ + i] = bq[i];   lb[BI_BK + i] = bk[i];
    lb[BI_BV + i] = bv[i];   lb[BI_BO + i] = bo[i];
    lb[BI_FB2 + i] = fb2[i]; lb[BI_BKP + i] = bkp[i];
    lb[BI_L1G + i] = ln1g[i]; lb[BI_L1B + i] = ln1b[i];
    lb[BI_L2G + i] = ln2g[i]; lb[BI_L2B + i] = ln2b[i];
  }
  for (int i = tid; i < DFF; i += NTHREADS) lb[BI_FB1 + i] = fb1[i];

  // ---- stage 1: e[t][j] = relu(x_t * veW1[j] + veB1[j]), rows padded to 112
  for (int i = tid; i < MPAD * 64; i += NTHREADS) {
    int t = i >> 6, j = i & 63;
    float v = 0.f;
    if (t < NTOK) v = fmaxf(x[b * NTOK + t] * veW1[j] + veB1[j], 0.f);
    e16[i] = (half_t)v;
  }
  __syncthreads();

  // ---- stage 2: h = e @ veW2 + posq (branch-free; pads come out exactly 0)
  gemm16<7, 8, 2>(e16, 64, BWS(OFF_VE, 8), [&](int r, int c, float v) {
    float y = v + posq[r * DMODEL + c];
    h32[r * DMODEL + c] = y;
    h16[r * DMODEL + c] = (half_t)y;
  });
  __syncthreads();

  // ---- stage 3: Q,K,V = h@W + b  (V stored transposed for attn@V B-frags)
  for (int i = tid; i < DMODEL * 16; i += NTHREADS) {  // zero v^T K-pad s=112..127
    int d = i >> 4, s = 112 + (i & 15);
    vt16[d * DMODEL + s] = (half_t)0.f;
  }
  gemm16<7, 8, 4>(h16, DMODEL, BWS(OFF_WQ, 8), [&](int r, int c, float v) {
    q16[r * DMODEL + c] = (half_t)(v + lb[BI_BQ + c]);
  });
  gemm16<7, 8, 4>(h16, DMODEL, BWS(OFF_WK, 8), [&](int r, int c, float v) {
    k16[r * DMODEL + c] = (half_t)(v + lb[BI_BK + c]);
  });
  gemm16<7, 8, 4>(h16, DMODEL, BWS(OFF_WV, 8), [&](int r, int c, float v) {
    vt16[c * DMODEL + r] = (half_t)(v + lb[BI_BV + c]);
  });
  __syncthreads();

  // ---- stage 4: attention per head (d_head=64)
  for (int h = 0; h < 2; ++h) {
    const int ho = h * 64;
    // S = Q_h @ K_h^T * (1/sqrt(64))
    gemm16<7, 7, 2>(q16 + ho, DMODEL, BTR(k16 + ho, DMODEL),
                    [&](int r, int c, float v) {
                      sc32[r * DMODEL + c] = v * 0.125f;
                    });
    __syncthreads();
    // row softmax over valid tokens only; zero all pads
    if (tid < MPAD) {
      half_t* arow = at16 + tid * DMODEL;
      if (tid < NTOK) {
        const float* srow = sc32 + tid * DMODEL;
        float m = -INFINITY;
        for (int s = 0; s < NTOK; ++s) m = fmaxf(m, srow[s]);
        float sum = 0.f;
        for (int s = 0; s < NTOK; ++s) sum += __expf(srow[s] - m);
        float inv = 1.f / sum;
        for (int s = 0; s < NTOK; ++s) arow[s] = (half_t)(__expf(srow[s] - m) * inv);
        for (int s = NTOK; s < DMODEL; ++s) arow[s] = (half_t)0.f;
      } else {
        for (int s = 0; s < DMODEL; ++s) arow[s] = (half_t)0.f;
      }
    }
    __syncthreads();
    // out_h = attn @ V_h   (B from V^T rows d = ho..ho+63)
    gemm16<7, 4, 4>(at16, DMODEL, BTR(vt16 + ho * DMODEL, DMODEL),
                    [&](int r, int c, float v) {
                      ao16[r * DMODEL + ho + c] = (half_t)v;
                    });
    __syncthreads();
  }

  // ---- stage 5: out2 = attn_out @ Wo + bo
  gemm16<7, 8, 4>(ao16, DMODEL, BWS(OFF_WO, 8), [&](int r, int c, float v) {
    sc32[r * DMODEL + c] = v + lb[BI_BO + c];
  });
  __syncthreads();

  // ---- stage 6: h = LN1(h + out2)
  if (tid < NTOK) {
    float* row = h32 + tid * DMODEL;
    const float* ad = sc32 + tid * DMODEL;
    float mu = 0.f;
    for (int d = 0; d < DMODEL; ++d) mu += row[d] + ad[d];
    mu *= (1.f / DMODEL);
    float var = 0.f;
    for (int d = 0; d < DMODEL; ++d) {
      float t = row[d] + ad[d] - mu;
      var += t * t;
    }
    float inv = rsqrtf(var * (1.f / DMODEL) + 1e-5f);
    for (int d = 0; d < DMODEL; ++d) {
      float y = (row[d] + ad[d] - mu) * inv * lb[BI_L1G + d] + lb[BI_L1B + d];
      row[d] = y;
      h16[tid * DMODEL + d] = (half_t)y;
    }
  }
  __syncthreads();

  // ---- stage 7: f1 = relu(h @ W1 + b1)   [112][256]
  gemm16<7, 16, 4>(h16, DMODEL, BWS(OFF_W1, 16), [&](int r, int c, float v) {
    f116[r * DFF + c] = (half_t)fmaxf(v + lb[BI_FB1 + c], 0.f);
  });
  __syncthreads();

  // ---- stage 8: f2 = f1 @ W2 + b2
  gemm16<7, 8, 8>(f116, DFF, BWS(OFF_W2, 8), [&](int r, int c, float v) {
    sc32[r * DMODEL + c] = v + lb[BI_FB2 + c];
  });
  __syncthreads();

  // ---- stage 9: h = LN2(h + f2)
  if (tid < NTOK) {
    float* row = h32 + tid * DMODEL;
    const float* ad = sc32 + tid * DMODEL;
    float mu = 0.f;
    for (int d = 0; d < DMODEL; ++d) mu += row[d] + ad[d];
    mu *= (1.f / DMODEL);
    float var = 0.f;
    for (int d = 0; d < DMODEL; ++d) {
      float t = row[d] + ad[d] - mu;
      var += t * t;
    }
    float inv = rsqrtf(var * (1.f / DMODEL) + 1e-5f);
    for (int d = 0; d < DMODEL; ++d) {
      float y = (row[d] + ad[d] - mu) * inv * lb[BI_L2G + d] + lb[BI_L2B + d];
      row[d] = y;
      h16[tid * DMODEL + d] = (half_t)y;
    }
  }
  __syncthreads();

  // ---- stage 10: keys = h @ Wkp + bkp (f16, row-major -> B^T source)
  gemm16<7, 8, 4>(h16, DMODEL, BWS(OFF_WKP, 8), [&](int r, int c, float v) {
    key16[r * DMODEL + c] = (half_t)(v + lb[BI_BKP + c]);
  });
  __syncthreads();

  // ---- stage 11: S2[r][n] = queries[r] . keys[n]   (A from global ws)
  gemm16<7, 7, 4>(ws + OFF_QRY, DMODEL, BTR(key16, DMODEL),
                  [&](int r, int c, float v) { sc32[r * DMODEL + c] = v; });
  if (tid < DMODEL) pick[tid] = 0x7fffffff;
  __syncthreads();

  // ---- stage 12: sequential greedy pointer decode on wave 0
  if (tid < 32) {
    const int lane = tid;
    float msk[4] = {0.f, 0.f, 0.f, 0.f};
    for (int r = 0; r < NTOK; ++r) {
      float best = -INFINITY;
      int bi = 0x7fffffff;
      for (int j = 0; j < 4; ++j) {
        int n = lane + 32 * j;
        if (n < NTOK) {
          float v = sc32[r * DMODEL + n] + msk[j];
          if (v > best || (v == best && n < bi)) { best = v; bi = n; }
        }
      }
      for (int off = 16; off; off >>= 1) {
        float ov = __shfl_xor(best, off, 32);
        int oi = __shfl_xor(bi, off, 32);
        if (ov > best || (ov == best && oi < bi)) { best = ov; bi = oi; }
      }
      if (lane == 0) pick[bi] = r;            // bi uniform across wave
      if ((bi & 31) == lane) msk[bi >> 5] = -INFINITY;
    }
  }
  __syncthreads();

  // ---- stage 13: out[b][r][n] = S2[r][n] + (picked before step r ? -inf : 0)
  float* outb = out + (size_t)b * (NTOK * NTOK);
  for (int r = 0; r < NTOK; ++r) {
    for (int n = tid; n < NTOK; n += NTHREADS) {  // tid<100 active; coalesced
      float pen = (pick[n] < r) ? -INFINITY : 0.f;
      outb[r * NTOK + n] = sc32[r * DMODEL + n] + pen;
    }
  }
}

// ---------------- host entry ----------------
extern "C" void kernel_launch(void* const* d_in, const int* in_sizes, int n_in,
                              void* d_out, int out_size, void* d_ws,
                              size_t ws_size, hipStream_t stream) {
  (void)in_sizes; (void)n_in; (void)out_size; (void)ws_size;
  const float* x    = (const float*)d_in[0];
  const float* veW1 = (const float*)d_in[1];
  const float* veB1 = (const float*)d_in[2];
  const float* veW2 = (const float*)d_in[3];
  const float* veB2 = (const float*)d_in[4];
  const float* posE = (const float*)d_in[5];
  const float* rankE= (const float*)d_in[6];
  const float* Wq   = (const float*)d_in[7];
  const float* bq   = (const float*)d_in[8];
  const float* Wk   = (const float*)d_in[9];
  const float* bk   = (const float*)d_in[10];
  const float* Wv   = (const float*)d_in[11];
  const float* bv   = (const float*)d_in[12];
  const float* Wo   = (const float*)d_in[13];
  const float* bo   = (const float*)d_in[14];
  const float* W1   = (const float*)d_in[15];
  const float* fb1  = (const float*)d_in[16];
  const float* W2   = (const float*)d_in[17];
  const float* fb2  = (const float*)d_in[18];
  const float* ln1g = (const float*)d_in[19];
  const float* ln1b = (const float*)d_in[20];
  const float* ln2g = (const float*)d_in[21];
  const float* ln2b = (const float*)d_in[22];
  const float* Wqp  = (const float*)d_in[23];
  const float* bqp  = (const float*)d_in[24];
  const float* Wkp  = (const float*)d_in[25];
  const float* bkp  = (const float*)d_in[26];
  half_t* ws = (half_t*)d_ws;
  float* outp = (float*)d_out;

  // allow >64KB dynamic LDS (no-op if already permitted); not a stream op
  hipFuncSetAttribute((const void*)pds_main,
                      hipFuncAttributeMaxDynamicSharedMemorySize, LDS_BYTES);

  pds_prep<<<dim3(256), dim3(256), 0, stream>>>(rankE, Wqp, bqp, veW2, Wq, Wk,
                                                Wv, Wo, W1, W2, Wkp, veB2, posE,
                                                ws);
  pds_main<<<dim3(B_SZ), dim3(NTHREADS), LDS_BYTES, stream>>>(
      x, veW1, veB1, bq, bk, bv, bo, fb1, fb2, ln1g, ln1b, ln2g, ln2b, bkp, ws,
      outp);
}
